// HierarchicalSparseExpertLayer_89197880804001
// MI455X (gfx1250) — matrix operand order repaired and checked by reading, not compiled, not run on hardware
//
#include <hip/hip_runtime.h>
#include <hip/hip_bf16.h>
#include <math.h>

// Problem constants (from reference)
#define N_TOK 8192
#define DIM   1024
#define NEXP  64
#define RANK  64
#define NGRP  8
#define MPG   8
#define TOPK  2

typedef __attribute__((ext_vector_type(16))) __bf16 v16bf;
typedef __attribute__((ext_vector_type(8)))  float  v8f;

#define HS_STRIDE 1032   // 1024 + 8 bf16 pad (break LDS bank alignment)
#define TB_STRIDE 72     // 64 + 8 bf16 pad

// CDNA5 async global->LDS path (guarded so both toolchains compile)
#if defined(__has_builtin)
#if __has_builtin(__builtin_amdgcn_global_load_async_to_lds_b128)
#define HAVE_ASYNC_LDS 1
#endif
#if __has_builtin(__builtin_amdgcn_s_wait_asynccnt)
#define HAVE_WAIT_ASYNC 1
#endif
#endif

#if defined(HAVE_ASYNC_LDS)
// exact parameter types per hipcc diagnostic:
//   param0: int __attribute__((vector_size(16))) __device__(AS1) *
//   param1: LDS-side pointer (AS3)
typedef int v4i_t __attribute__((vector_size(16)));
typedef __attribute__((address_space(1))) v4i_t* g_v4i_p;
typedef __attribute__((address_space(3))) v4i_t* l_v4i_p;
#endif

// ---------------------------------------------------------------------------
// Kernel 0: zero output + per-expert counters
// ---------------------------------------------------------------------------
__global__ void zero_kernel(float* __restrict__ out, int n4, int* __restrict__ counts) {
  int i = blockIdx.x * blockDim.x + threadIdx.x;
  if (i < n4) {
    float4 z = make_float4(0.f, 0.f, 0.f, 0.f);
    ((float4*)out)[i] = z;
  }
  if (i < NEXP) counts[i] = 0;
}

// ---------------------------------------------------------------------------
// Kernel 1: router (group argmax -> within-group top-2 -> softmax gate)
// ---------------------------------------------------------------------------
__global__ void router_kernel(const float* __restrict__ h,
                              const float* __restrict__ Wg,
                              const float* __restrict__ bg,
                              const float* __restrict__ Wloc,
                              float* __restrict__ eid_out,   // (N,2) as float
                              float* __restrict__ gate_out,  // (N,2)
                              float* __restrict__ gidx_out,  // (N,)
                              int*   __restrict__ tok_eid,   // ws (N,2)
                              float* __restrict__ tok_gate,  // ws (N,2)
                              int*   __restrict__ counts) {  // ws (E,)
  int n = blockIdx.x * blockDim.x + threadIdx.x;
  if (n >= N_TOK) return;
  const float* hrow = h + (size_t)n * DIM;

  float acc[NGRP];
#pragma unroll
  for (int g = 0; g < NGRP; ++g) acc[g] = bg[g];
  for (int d = 0; d < DIM; d += 4) {
    float4 hv = *(const float4*)(hrow + d);
#pragma unroll
    for (int g = 0; g < NGRP; ++g) {
      acc[g] += hv.x * Wg[(d + 0) * NGRP + g];
      acc[g] += hv.y * Wg[(d + 1) * NGRP + g];
      acc[g] += hv.z * Wg[(d + 2) * NGRP + g];
      acc[g] += hv.w * Wg[(d + 3) * NGRP + g];
    }
  }
  int gi = 0; float gv = acc[0];
#pragma unroll
  for (int g = 1; g < NGRP; ++g) if (acc[g] > gv) { gv = acc[g]; gi = g; }

  float mac[MPG];
#pragma unroll
  for (int m = 0; m < MPG; ++m) mac[m] = 0.f;
  const float* wl = Wloc + (size_t)gi * DIM * MPG;
  for (int d = 0; d < DIM; d += 4) {
    float4 hv = *(const float4*)(hrow + d);
#pragma unroll
    for (int m = 0; m < MPG; ++m) {
      mac[m] += hv.x * wl[(d + 0) * MPG + m];
      mac[m] += hv.y * wl[(d + 1) * MPG + m];
      mac[m] += hv.z * wl[(d + 2) * MPG + m];
      mac[m] += hv.w * wl[(d + 3) * MPG + m];
    }
  }
  int i1 = 0; float v1 = mac[0];
#pragma unroll
  for (int m = 1; m < MPG; ++m) if (mac[m] > v1) { v1 = mac[m]; i1 = m; }
  int i2 = -1; float v2 = -3.4e38f;
#pragma unroll
  for (int m = 0; m < MPG; ++m) if (m != i1 && mac[m] > v2) { v2 = mac[m]; i2 = m; }

  float b = __expf(v2 - v1);
  float g1 = 1.f / (1.f + b);
  float g2 = b * g1;

  int e1 = gi * MPG + i1, e2 = gi * MPG + i2;
  eid_out[n * 2 + 0] = (float)e1;
  eid_out[n * 2 + 1] = (float)e2;
  gate_out[n * 2 + 0] = g1;
  gate_out[n * 2 + 1] = g2;
  gidx_out[n] = (float)gi;

  tok_eid[n * 2 + 0] = e1; tok_eid[n * 2 + 1] = e2;
  tok_gate[n * 2 + 0] = g1; tok_gate[n * 2 + 1] = g2;
  atomicAdd(&counts[e1], 1);
  atomicAdd(&counts[e2], 1);
}

// ---------------------------------------------------------------------------
// Kernel 2: tiny scan over 64 experts -> list offsets + 16-row tile offsets
// ---------------------------------------------------------------------------
__global__ void scan_kernel(const int* __restrict__ counts,
                            int* __restrict__ offsets,
                            int* __restrict__ tileOff,
                            int* __restrict__ cursor) {
  if (threadIdx.x == 0) {
    int off = 0, toff = 0;
    for (int e = 0; e < NEXP; ++e) {
      offsets[e] = off; tileOff[e] = toff;
      off  += counts[e];
      toff += (counts[e] + 15) >> 4;
    }
    offsets[NEXP] = off; tileOff[NEXP] = toff;
  }
  if (threadIdx.x < NEXP) cursor[threadIdx.x] = 0;
}

// ---------------------------------------------------------------------------
// Kernel 3: scatter (token, gate) into per-expert assignment lists
// ---------------------------------------------------------------------------
__global__ void scatter_kernel(const int* __restrict__ tok_eid,
                               const float* __restrict__ tok_gate,
                               const int* __restrict__ offsets,
                               int* __restrict__ cursor,
                               int* __restrict__ a_tok,
                               float* __restrict__ a_gate) {
  int n = blockIdx.x * blockDim.x + threadIdx.x;
  if (n >= N_TOK) return;
#pragma unroll
  for (int j = 0; j < TOPK; ++j) {
    int e = tok_eid[n * 2 + j];
    int pos = offsets[e] + atomicAdd(&cursor[e], 1);
    a_tok[pos]  = n;
    a_gate[pos] = tok_gate[n * 2 + j];
  }
}

// ---------------------------------------------------------------------------
// WMMA fragment helpers (contiguous k order used here, wave32)
// A 16x32 bf16: lane L -> row m=L%16, elem i -> k=(L/16)*16+i
// B 32x16 bf16: lane L -> col n=L%16, elem i -> k=(L/16)*16+i
// C 16x16 f32 : lane L -> col n=L%16, vgpr j -> row m=j+8*(L/16)
// ---------------------------------------------------------------------------
__device__ inline v16bf load_a_lds(const __bf16* base, int stride, int lane, int k0) {
  int m = lane & 15;
  int kb = (lane >> 4) * 16;
  const __bf16* row = base + m * stride + k0 + kb;
  v16bf a;
#pragma unroll
  for (int i = 0; i < 16; ++i) a[i] = row[i];
  return a;
}

// producer-side: stage one contiguous 8KB W1 K-chunk (32 rows x 64 cols f32)
// into LDS. Async (ASYNCcnt-tracked b128 transfers) when available.
__device__ inline void stage_chunk(const float* __restrict__ src,
                                   float* __restrict__ dst, int ptid) {
#if defined(HAVE_ASYNC_LDS)
#pragma unroll
  for (int t = 0; t < 4; ++t) {
    int idx = ptid + t * 128;                 // 512 x 16B over 128 threads
    __builtin_amdgcn_global_load_async_to_lds_b128(
        (g_v4i_p)(src + 4 * idx), (l_v4i_p)(dst + 4 * idx), 0, 0);
  }
#else
#pragma unroll
  for (int t = 0; t < 4; ++t) {
    int idx = ptid + t * 128;
    ((float4*)dst)[idx] = ((const float4*)src)[idx];
  }
#endif
}

__device__ inline void wait_async() {
#if defined(HAVE_ASYNC_LDS)
#if defined(HAVE_WAIT_ASYNC)
  __builtin_amdgcn_s_wait_asynccnt(0);
#else
  asm volatile("s_wait_asynccnt 0x0" ::: "memory");
#endif
#endif
}

// ---------------------------------------------------------------------------
// Kernel 4: expert compute. One block = one (expert, 16-token tile).
// Warp-specialized: waves 0-3 = WMMA consumers, waves 4-7 = async DMA
// producers (double-buffered W1 stage + W2 prefetch).
//   tmp(16x64) = relu(h_tile(16x1024) @ W1[e])      [bf16 WMMA, f32 acc]
//   out_tile(16x1024) += gate * (tmp @ W2[e])       [bf16 WMMA, f32 acc]
// ---------------------------------------------------------------------------
__global__ void __launch_bounds__(256)
expert_kernel(const float* __restrict__ h,
              const float* __restrict__ W1,
              const float* __restrict__ W2,
              const int* __restrict__ a_tok,
              const float* __restrict__ a_gate,
              const int* __restrict__ offsets,
              const int* __restrict__ tileOff,
              float* __restrict__ out) {
  __shared__ __bf16 hs[16 * HS_STRIDE];        // gathered tokens, bf16 (~33 KB)
  __shared__ float  w1buf[2][32 * RANK];       // double-buffered W1 chunks (16 KB)
  __shared__ __bf16 tmpb[16 * TB_STRIDE];      // relu(tmp) as bf16 (~2.3 KB)
  __shared__ float  gates[16];
  __shared__ int    toks[16];
  __shared__ int    meta[3];

  if (threadIdx.x == 0) {
    int b = blockIdx.x;
    if (b >= tileOff[NEXP]) { meta[0] = -1; }
    else {
      int e = 0;
      while (tileOff[e + 1] <= b) ++e;         // 64 entries, scalar scan
      meta[0] = e;
      meta[1] = (b - tileOff[e]) * 16;
      meta[2] = offsets[e];
    }
  }
  __syncthreads();
  int e = meta[0];
  if (e < 0) return;
  int rowBase = meta[1], listStart = meta[2];

  if (threadIdx.x < 16) {
    int count = offsets[e + 1] - listStart;
    int idx = rowBase + (int)threadIdx.x;
    int cl = idx < count ? idx : count - 1;    // clamp padding rows
    toks[threadIdx.x]  = a_tok[listStart + cl];
    gates[threadIdx.x] = (idx < count) ? a_gate[listStart + cl] : 0.0f;
  }
  __syncthreads();

  // stage 16 gathered h rows -> LDS bf16 (coalesced along D)
  for (int idx = threadIdx.x; idx < 16 * DIM; idx += 256) {
    int m = idx >> 10, col = idx & (DIM - 1);
    hs[m * HS_STRIDE + col] = (__bf16)h[(size_t)toks[m] * DIM + col];
  }
  __syncthreads();

  int wave = threadIdx.x >> 5, lane = threadIdx.x & 31;
  const float* w1e = W1 + (size_t)e * DIM * RANK;
  const float* w2e = W2 + (size_t)e * RANK * DIM;

  // ---- phase 1: tmp = h_tile @ W1[e] ----
  // consumers: wave w in 0..3 owns N-tile w, full K loop, acc in VGPRs.
  // producers: waves 4..7 async-stage W1 chunk kc+1 while kc is consumed,
  //            and trickle global_prefetch over the W2 stream for phase 2.
  int r = (wave & 3) * 16 + (lane & 15);
  v8f c = {};
  int ptid = (int)threadIdx.x - 128;
  if (wave >= 4) {
    stage_chunk(w1e, w1buf[0], ptid);
    wait_async();
  }
  __syncthreads();

  for (int kc = 0; kc < 32; ++kc) {
    if (wave >= 4) {
      if (kc + 1 < 32)
        stage_chunk(w1e + (size_t)(kc + 1) * 32 * RANK, w1buf[(kc + 1) & 1], ptid);
      // prefetch W2[e] for phase 2: 2048 x 128B lines spread over 32 iters
      if ((ptid & 1) == 0) {
        int line = kc * 64 + (ptid >> 1);
        __builtin_prefetch(w2e + (size_t)line * 32, 0, 3);
      }
      wait_async();
    } else {
      int k0 = kc * 32;
      v16bf a = load_a_lds(hs, HS_STRIDE, lane, k0);
      const float* buf = w1buf[kc & 1];
      int kb = (lane >> 4) * 16;
      v16bf b;
#pragma unroll
      for (int i = 0; i < 16; ++i) b[i] = (__bf16)buf[(kb + i) * RANK + r];
      c = __builtin_amdgcn_wmma_f32_16x16x32_bf16(false, a, false, b,
                                                  (short)0, c, false, false);
    }
    __syncthreads();
  }

  // consumers publish relu(tmp) as bf16 (no reduce pass needed: full-K acc)
  if (wave < 4) {
#pragma unroll
    for (int j = 0; j < 8; ++j) {
      int m = j + ((lane >> 4) << 3);
      tmpb[m * TB_STRIDE + r] = (__bf16)fmaxf(c[j], 0.0f);
    }
  }
  __syncthreads();

  // ---- phase 2: out_tile = (gate * tmp) @ W2[e], 64 N-tiles over 8 waves ----
  for (int nt = wave; nt < DIM / 16; nt += 8) {
    int dcol = nt * 16 + (lane & 15);
    v8f c2 = {};
#pragma unroll
    for (int kk = 0; kk < 2; ++kk) {
      int k0 = kk * 32;
      v16bf a = load_a_lds(tmpb, TB_STRIDE, lane, k0);
      int kb = k0 + ((lane >> 4) * 16);
      v16bf b;
#pragma unroll
      for (int i = 0; i < 16; ++i) b[i] = (__bf16)w2e[(size_t)(kb + i) * DIM + dcol];
      c2 = __builtin_amdgcn_wmma_f32_16x16x32_bf16(false, a, false, b,
                                                   (short)0, c2, false, false);
    }
#pragma unroll
    for (int j = 0; j < 8; ++j) {
      int m = j + ((lane >> 4) << 3);
      atomicAdd(&out[(size_t)toks[m] * DIM + dcol], c2[j] * gates[m]);
    }
  }
}

// ---------------------------------------------------------------------------
// launcher: out layout = [out NxD | expert_ids Nx2 | gate Nx2 | group_idx N]
// (int outputs written as float values). k input (d_in[6]) is fixed at 2.
// ws usage: ~264 KB of int/float scratch.
// ---------------------------------------------------------------------------
extern "C" void kernel_launch(void* const* d_in, const int* in_sizes, int n_in,
                              void* d_out, int out_size, void* d_ws, size_t ws_size,
                              hipStream_t stream) {
  const float* h    = (const float*)d_in[0];
  const float* Wg   = (const float*)d_in[1];
  const float* bg   = (const float*)d_in[2];
  const float* Wloc = (const float*)d_in[3];
  const float* W1   = (const float*)d_in[4];
  const float* W2   = (const float*)d_in[5];

  float* out      = (float*)d_out;
  float* eid_out  = out + (size_t)N_TOK * DIM;
  float* gate_out = eid_out + N_TOK * TOPK;
  float* gidx_out = gate_out + N_TOK * TOPK;

  int*   counts   = (int*)d_ws;              // 64
  int*   offsets  = counts + 64;             // 65
  int*   tileOff  = offsets + 65;            // 65
  int*   cursor   = tileOff + 65;            // 64
  int*   tok_eid  = (int*)d_ws + 512;        // N*2
  float* tok_gate = (float*)((int*)d_ws + 512 + N_TOK * TOPK);
  int*   a_tok    = (int*)d_ws + 512 + 2 * N_TOK * TOPK;
  float* a_gate   = (float*)((int*)d_ws + 512 + 3 * N_TOK * TOPK);

  int n4 = (N_TOK * DIM) / 4;
  zero_kernel<<<(n4 + 255) / 256, 256, 0, stream>>>(out, n4, counts);
  router_kernel<<<N_TOK / 256, 256, 0, stream>>>(h, Wg, bg, Wloc, eid_out, gate_out,
                                                 gidx_out, tok_eid, tok_gate, counts);
  scan_kernel<<<1, 64, 0, stream>>>(counts, offsets, tileOff, cursor);
  scatter_kernel<<<N_TOK / 256, 256, 0, stream>>>(tok_eid, tok_gate, offsets, cursor,
                                                  a_tok, a_gate);
  int maxTiles = (N_TOK * TOPK) / 16 + NEXP;  // worst-case partial tiles
  expert_kernel<<<maxTiles, 256, 0, stream>>>(h, W1, W2, a_tok, a_gate,
                                              offsets, tileOff, out);
}
